// DANAttention_78391743087021
// MI455X (gfx1250) — compile-verified, matrix operand order checked by reading, not run on hardware
//
#include <hip/hip_runtime.h>
#include <hip/hip_bf16.h>
#include <math.h>

// Problem constants (from reference): B=2, S=2048, E=768, H=12, D=64, TB=128
#define BB    2
#define SS    2048
#define EE    768
#define HH    12
#define DD    64
#define TBX   128
#define BH    (BB*HH)          // 24
#define MROWS (BB*SS)          // 4096
#define LDP   2052             // padded LDS row pitch (floats) to dodge bank conflicts

typedef __attribute__((ext_vector_type(16))) __bf16 v16bf;
typedef __attribute__((ext_vector_type(8)))  float  v8f;

// ---------------------------------------------------------------------------
// WMMA bf16 fragment loaders (wave32, V_WMMA_*_16X16X32_BF16 layouts).
// A 16x32 (MxK) 16-bit: lanes 0-15 -> M=lane, K in {0..7,16..23};
//                       lanes 16-31 -> M=lane-16, K in {8..15,24..31};
//                       vector elements (2j,2j+1) sit in VGPR j.
// B 32x16 (KxN): symmetric layout with N <- lane&15.
// C/D 16x16 f32: VGPR r -> M = r + 8*(lane>>4), N = lane&15.
// ---------------------------------------------------------------------------
__device__ __forceinline__ v16bf frag_row_f32(const float* __restrict__ base,
                                              int ld, int row0, int kb, int lane) {
  const float* p = base + (size_t)(row0 + (lane & 15)) * ld + kb + ((lane >> 4) << 3);
  v16bf f;
#pragma unroll
  for (int j = 0; j < 4; ++j) {
    f[2*j]         = (__bf16)p[2*j];
    f[2*j + 1]     = (__bf16)p[2*j + 1];
    f[8 + 2*j]     = (__bf16)p[16 + 2*j];
    f[8 + 2*j + 1] = (__bf16)p[16 + 2*j + 1];
  }
  return f;
}

__device__ __forceinline__ v16bf frag_row_bf16(const __bf16* __restrict__ base,
                                               int ld, int row0, int kb, int lane) {
  const __bf16* p = base + (size_t)(row0 + (lane & 15)) * ld + kb + ((lane >> 4) << 3);
  v16bf f;
#pragma unroll
  for (int j = 0; j < 4; ++j) {
    f[2*j]         = p[2*j];
    f[2*j + 1]     = p[2*j + 1];
    f[8 + 2*j]     = p[16 + 2*j];
    f[8 + 2*j + 1] = p[16 + 2*j + 1];
  }
  return f;
}

__device__ __forceinline__ v8f wmma_bf16(v16bf a, v16bf b, v8f c) {
  return __builtin_amdgcn_wmma_f32_16x16x32_bf16(false, a, false, b, (short)0, c,
                                                 false, false);
}

// ---------------------------------------------------------------------------
// Kernel 0: tb[b,h] = task_bias[b,:] @ Wtb[h,:] + btb[h]
// ---------------------------------------------------------------------------
__global__ void k_taskbias(const float* __restrict__ task_bias,
                           const float* __restrict__ Wtb,
                           const float* __restrict__ btb,
                           float* __restrict__ tb) {
  int i = threadIdx.x;
  if (i >= BH) return;
  int b = i / HH, h = i % HH;
  float s = btb[h];
  for (int t = 0; t < TBX; ++t) s += task_bias[b * TBX + t] * Wtb[h * TBX + t];
  tb[i] = s;
}

// ---------------------------------------------------------------------------
// Kernel 1: projection  P = X @ W^T + bias ; fp32 in, bf16 out.
//   TRANS=false: out layout [B,H,S,D]   (Q, K)
//   TRANS=true : out layout [B,H,D,S]   (V, so attn@V B-frags are contiguous)
// One wave per 16x16 tile; 8 waves/block; 12288 tiles.
// ---------------------------------------------------------------------------
template <bool TRANS>
__global__ void k_proj(const float* __restrict__ X, const float* __restrict__ W,
                       const float* __restrict__ bias, __bf16* __restrict__ out) {
  const int lane = threadIdx.x & 31;
  const int tile = blockIdx.x * 8 + (threadIdx.x >> 5);
  const int NT = EE / 16;               // 48
  const int mt = tile / NT, nt = tile % NT;
  const int mb = mt * 16, nb = nt * 16;

  v8f acc = {};
#pragma unroll 4
  for (int kb = 0; kb < EE; kb += 32) {
    v16bf a = frag_row_f32(X, EE, mb, kb, lane);
    v16bf b = frag_row_f32(W, EE, nb, kb, lane);   // rows of W == cols of W^T
    acc = wmma_bf16(a, b, acc);
  }
  const int n  = nb + (lane & 15);
  const float bv = bias[n];
  const int h = n >> 6, d = n & 63;
#pragma unroll
  for (int r = 0; r < 8; ++r) {
    int m = mb + r + ((lane >> 4) << 3);
    int bi = m >> 11, s = m & (SS - 1);
    if (TRANS)
      out[(((size_t)bi * HH + h) * DD + d) * SS + s] = (__bf16)(acc[r] + bv);
    else
      out[(((size_t)bi * HH + h) * SS + s) * DD + d] = (__bf16)(acc[r] + bv);
  }
}

// ---------------------------------------------------------------------------
// Kernel 2 (fused): per (b,h,qtile) -> scores in LDS -> softmax in LDS
//                   -> one NT write of attn -> ctx from LDS.
// Block: 256 threads (8 waves). Dynamic LDS: sc[16][LDP] fp32 + cpart[4][256].
// ---------------------------------------------------------------------------
__global__ void k_attn_fused(const __bf16* __restrict__ Q,
                             const __bf16* __restrict__ K,
                             const __bf16* __restrict__ Vt,   // [B,H,D,S]
                             const float* __restrict__ tb,
                             const int* __restrict__ mask,
                             float* __restrict__ attn,
                             __bf16* __restrict__ ctx) {
  extern __shared__ char smem[];
  float* sc    = (float*)smem;                               // [16][LDP]
  float* cpart = (float*)(smem + (size_t)16 * LDP * sizeof(float)); // [4][256]

  const int lane = threadIdx.x & 31;
  const int wave = threadIdx.x >> 5;
  const int qt   = blockIdx.x & (SS / 16 - 1);               // 0..127
  const int bh   = blockIdx.x >> 7;                          // 0..23
  const int bidx = bh / HH, h = bh % HH;

  const __bf16* Qh = Q  + (size_t)bh * SS * DD;
  const __bf16* Kh = K  + (size_t)bh * SS * DD;
  const __bf16* Vh = Vt + (size_t)bh * DD * SS;

  // ---- Phase 1: scores strip [16][2048] -> LDS --------------------------
  const v16bf qa0 = frag_row_bf16(Qh, DD, qt * 16, 0,  lane);
  const v16bf qa1 = frag_row_bf16(Qh, DD, qt * 16, 32, lane);
  const float bias = tb[bh];
#pragma unroll 2
  for (int ct = 0; ct < 16; ++ct) {
    const int kt = wave * 16 + ct;                           // column tile
    v16bf b0 = frag_row_bf16(Kh, DD, kt * 16, 0,  lane);     // B[k][n]=Kh[n][k]
    v16bf b1 = frag_row_bf16(Kh, DD, kt * 16, 32, lane);
    v8f acc = {};
    acc = wmma_bf16(qa0, b0, acc);
    acc = wmma_bf16(qa1, b1, acc);
    const int col  = kt * 16 + (lane & 15);
    const float mv = (mask[bidx * SS + col] == 0) ? -INFINITY : 0.0f;
#pragma unroll
    for (int r = 0; r < 8; ++r) {
      int row = r + ((lane >> 4) << 3);
      sc[row * LDP + col] = acc[r] * 0.125f + bias + mv;
    }
  }
  __syncthreads();

  // ---- Phase 2: softmax (2 rows per wave) + single NT write of attn -----
  float* arow_base = attn + ((size_t)bh * SS + (size_t)qt * 16) * SS;
#pragma unroll
  for (int rr = 0; rr < 2; ++rr) {
    const int row = wave * 2 + rr;
    float* r = sc + row * LDP;
    float mx = -INFINITY;
    for (int i = lane; i < SS; i += 32) mx = fmaxf(mx, r[i]);
#pragma unroll
    for (int o = 16; o > 0; o >>= 1) mx = fmaxf(mx, __shfl_xor(mx, o));
    float sum = 0.0f;
    for (int i = lane; i < SS; i += 32) {
      float e = __expf(r[i] - mx);
      r[i] = e;
      sum += e;
    }
#pragma unroll
    for (int o = 16; o > 0; o >>= 1) sum += __shfl_xor(sum, o);
    const float inv = 1.0f / sum;
    float* grow = arow_base + (size_t)row * SS;
    for (int i = lane; i < SS; i += 32) {
      float v = r[i] * inv;
      r[i] = v;
      __builtin_nontemporal_store(v, grow + i);              // streaming write
    }
  }
  __syncthreads();

  // ---- Phase 3: ctx tile = attn(LDS) @ V ; 4 N-tiles x 2 K-halves -------
  const int nt = wave & 3;              // d-tile 0..3
  const int kh = wave >> 2;             // K half 0..1
  v8f acc = {};
  for (int kb = kh * (SS / 2); kb < (kh + 1) * (SS / 2); kb += 32) {
    v16bf a = frag_row_f32(sc, LDP, 0, kb, lane);            // LDS -> bf16
    v16bf b = frag_row_bf16(Vh, SS, nt * 16, kb, lane);      // B[k][n]=Vt[n][k]
    acc = wmma_bf16(a, b, acc);
  }
  if (kh == 1) {
#pragma unroll
    for (int r = 0; r < 8; ++r) cpart[nt * 256 + r * 32 + lane] = acc[r];
  }
  __syncthreads();
  if (kh == 0) {
    const int d = nt * 16 + (lane & 15);
#pragma unroll
    for (int r = 0; r < 8; ++r) {
      float v = acc[r] + cpart[nt * 256 + r * 32 + lane];
      int q = qt * 16 + r + ((lane >> 4) << 3);
      ctx[((size_t)bidx * SS + q) * EE + h * DD + d] = (__bf16)v;
    }
  }
}

// ---------------------------------------------------------------------------
// Kernel 3: out = ctx @ Wo^T + bo ; bf16 ctx, fp32 Wo -> fp32 out [B,S,E].
// ---------------------------------------------------------------------------
__global__ void k_outproj(const __bf16* __restrict__ ctx, const float* __restrict__ Wo,
                          const float* __restrict__ bo, float* __restrict__ out) {
  const int lane = threadIdx.x & 31;
  const int tile = blockIdx.x * 8 + (threadIdx.x >> 5);
  const int NT = EE / 16;
  const int mt = tile / NT, nt = tile % NT;
  const int mb = mt * 16, nb = nt * 16;

  v8f acc = {};
#pragma unroll 4
  for (int kb = 0; kb < EE; kb += 32) {
    v16bf a = frag_row_bf16(ctx, EE, mb, kb, lane);
    v16bf b = frag_row_f32(Wo, EE, nb, kb, lane);
    acc = wmma_bf16(a, b, acc);
  }
  const int n = nb + (lane & 15);
  const float bv = bo[n];
#pragma unroll
  for (int r = 0; r < 8; ++r) {
    int m = mb + r + ((lane >> 4) << 3);
    out[(size_t)m * EE + n] = acc[r] + bv;
  }
}

// ---------------------------------------------------------------------------
// Host launcher
// ---------------------------------------------------------------------------
extern "C" void kernel_launch(void* const* d_in, const int* in_sizes, int n_in,
                              void* d_out, int out_size, void* d_ws, size_t ws_size,
                              hipStream_t stream) {
  (void)in_sizes; (void)n_in; (void)out_size; (void)ws_size;

  const float* query     = (const float*)d_in[0];
  const float* key       = (const float*)d_in[1];
  const float* value     = (const float*)d_in[2];
  const float* task_bias = (const float*)d_in[3];
  const int*   mask      = (const int*)  d_in[4];
  const float* Wq = (const float*)d_in[5];  const float* bq = (const float*)d_in[6];
  const float* Wk = (const float*)d_in[7];  const float* bk = (const float*)d_in[8];
  const float* Wv = (const float*)d_in[9];  const float* bv = (const float*)d_in[10];
  const float* Wtb = (const float*)d_in[11]; const float* btb = (const float*)d_in[12];
  const float* Wo = (const float*)d_in[13]; const float* bo = (const float*)d_in[14];

  float* out  = (float*)d_out;                       // [B,S,E]
  float* attn = out + (size_t)MROWS * EE;            // [B,H,S,S]

  // Workspace: Q,K bf16 [B,H,S,D]; V bf16 [B,H,D,S]; ctx bf16 [B,S,E]; tb fp32
  const size_t QKV = (size_t)BH * SS * DD;           // 3,145,728 elems
  __bf16* Qb = (__bf16*)d_ws;
  __bf16* Kb = Qb + QKV;
  __bf16* Vb = Kb + QKV;
  __bf16* Cb = Vb + QKV;
  float*  tb = (float*)(Cb + (size_t)MROWS * EE);

  k_taskbias<<<1, 32, 0, stream>>>(task_bias, Wtb, btb, tb);

  const int proj_tiles = (MROWS / 16) * (EE / 16);   // 12288
  k_proj<false><<<proj_tiles / 8, 256, 0, stream>>>(query, Wq, bq, Qb);
  k_proj<false><<<proj_tiles / 8, 256, 0, stream>>>(key,   Wk, bk, Kb);
  k_proj<true ><<<proj_tiles / 8, 256, 0, stream>>>(value, Wv, bv, Vb);

  // Fused attention: one block per (b,h,q-tile); 131.3KB scores + 4KB scratch LDS
  const size_t lds_bytes = (size_t)16 * LDP * sizeof(float) + 4 * 256 * sizeof(float);
  k_attn_fused<<<BH * (SS / 16), 256, lds_bytes, stream>>>(Qb, Kb, Vb, tb, mask,
                                                           attn, Cb);

  k_outproj<<<proj_tiles / 8, 256, 0, stream>>>(Cb, Wo, bo, out);
}
